// MemoryUpdateCell_5136780886051
// MI455X (gfx1250) — compile-verified, matrix operand order checked by reading, not run on hardware
//
#include <hip/hip_runtime.h>
#include <hip/hip_bf16.h>
#include <math.h>

typedef __bf16 bf16_t;
typedef __attribute__((ext_vector_type(16))) __bf16 v16bf;
typedef __attribute__((ext_vector_type(8)))  __bf16 v8bf;
typedef __attribute__((ext_vector_type(8)))  float  v8f;

#define B_TOTAL 4096
#define SEQ     50
#define HID     256
#define G3      768   // 3*HID
#define M_TILE  32    // batch rows per block
#define THREADS 256   // 8 wave32s

// ---------------------------------------------------------------------------
// Prep: convert fp32 weights to bf16 in workspace (runs every launch; cheap).
// ---------------------------------------------------------------------------
__global__ void convert_weights_kernel(const float* __restrict__ Wih,
                                       const float* __restrict__ Whh,
                                       bf16_t* __restrict__ WihB,
                                       bf16_t* __restrict__ WhhB) {
  int i = blockIdx.x * blockDim.x + threadIdx.x;
  if (i < G3 * HID) {
    WihB[i] = (bf16_t)Wih[i];
    WhhB[i] = (bf16_t)Whh[i];
  }
}

// ---------------------------------------------------------------------------
// Fragment loader for 16-bit WMMA operands (wave32).
// A 16x32 layout: lane L holds row M = L&15; lanes<16 carry K={0..7,16..23},
// lanes>=16 carry K={8..15,24..31}  ->  two contiguous 16B loads.
// Loading row-major W (row = gate n, 32 consecutive k) with the same pattern
// yields the B-operand fragment of W^T (lane = N column).
// ---------------------------------------------------------------------------
__device__ __forceinline__ v16bf load_frag(const bf16_t* base, int ld, int lane) {
  const int r    = lane & 15;
  const int half = lane >> 4;
  const bf16_t* p = base + r * ld + half * 8;
  v8bf lo = *(const v8bf*)(p);        // 16B: K offsets {0..7} (+8 for hi lanes)
  v8bf hi = *(const v8bf*)(p + 16);   // 16B: K offsets {16..23} (+8 for hi lanes)
  return __builtin_shufflevector(lo, hi, 0,1,2,3,4,5,6,7,8,9,10,11,12,13,14,15);
}

__device__ __forceinline__ float sigmoid_fast(float x) {
  return 1.0f / (1.0f + __expf(-x));
}

// ---------------------------------------------------------------------------
// Fused GRU gated scan. One block owns M_TILE batch rows for all 50 steps.
// h lives in LDS (f32 master + bf16 WMMA copy), double-buffered per step.
// ---------------------------------------------------------------------------
__launch_bounds__(THREADS, 2)
__global__ void gru_scan_kernel(const float* __restrict__ C,
                                const float* __restrict__ G,
                                const bf16_t* __restrict__ WihB,
                                const bf16_t* __restrict__ WhhB,
                                const float* __restrict__ b_ih,
                                const float* __restrict__ b_hh,
                                float* __restrict__ out) {
  __shared__ __align__(16) float  hf[2][M_TILE * HID];   // 2 x 32 KB
  __shared__ __align__(16) bf16_t hb[2][M_TILE * HID];   // 2 x 16 KB
  __shared__ __align__(16) bf16_t cs[M_TILE * HID];      // 16 KB

  const int tid  = threadIdx.x;
  const int lane = tid & 31;
  const int wave = tid >> 5;
  const int row0 = blockIdx.x * M_TILE;

  // 8 waves = 2 M-tiles x 4 N-groups (each N-group covers 4 hidden 16-tiles)
  const int mtile  = wave & 1;
  const int ngroup = wave >> 1;
  const int m0     = mtile * 16;

  // h0 = 0
  for (int i = tid; i < M_TILE * HID; i += THREADS) {
    hf[0][i] = 0.0f;
    hb[0][i] = (bf16_t)0.0f;
  }
  __syncthreads();

  int cur = 0;
  for (int s = 0; s < SEQ; ++s) {
    // Stage C[row0:row0+32, s, :] into LDS as bf16 (coalesced: row i, col tid)
    for (int i = 0; i < M_TILE; ++i) {
      float v = C[(size_t)(row0 + i) * (SEQ * HID) + (size_t)s * HID + tid];
      cs[i * HID + tid] = (bf16_t)v;
    }
    __syncthreads();

    const int nxt = cur ^ 1;
    const bf16_t* hcur = hb[cur];

    for (int t = 0; t < 4; ++t) {
      const int ntile = ngroup * 4 + t;
      const int n0    = ntile * 16;           // hidden column base (0..255)

      v8f aIR = {}, aIZ = {}, aIN = {}, aHR = {}, aHZ = {}, aHN = {};
      #pragma unroll
      for (int k = 0; k < 8; ++k) {           // K = 256 in chunks of 32
        const int kk = k * 32;
        v16bf fC  = load_frag(&cs[m0 * HID + kk],   HID, lane);
        v16bf fH  = load_frag(&hcur[m0 * HID + kk], HID, lane);
        v16bf bIR = load_frag(WihB + (size_t)(n0      ) * HID + kk, HID, lane);
        v16bf bIZ = load_frag(WihB + (size_t)(n0 + 256) * HID + kk, HID, lane);
        v16bf bIN = load_frag(WihB + (size_t)(n0 + 512) * HID + kk, HID, lane);
        v16bf bHR = load_frag(WhhB + (size_t)(n0      ) * HID + kk, HID, lane);
        v16bf bHZ = load_frag(WhhB + (size_t)(n0 + 256) * HID + kk, HID, lane);
        v16bf bHN = load_frag(WhhB + (size_t)(n0 + 512) * HID + kk, HID, lane);
        aIR = __builtin_amdgcn_wmma_f32_16x16x32_bf16(false, fC, false, bIR, (short)0, aIR, false, false);
        aIZ = __builtin_amdgcn_wmma_f32_16x16x32_bf16(false, fC, false, bIZ, (short)0, aIZ, false, false);
        aIN = __builtin_amdgcn_wmma_f32_16x16x32_bf16(false, fC, false, bIN, (short)0, aIN, false, false);
        aHR = __builtin_amdgcn_wmma_f32_16x16x32_bf16(false, fH, false, bHR, (short)0, aHR, false, false);
        aHZ = __builtin_amdgcn_wmma_f32_16x16x32_bf16(false, fH, false, bHZ, (short)0, aHZ, false, false);
        aHN = __builtin_amdgcn_wmma_f32_16x16x32_bf16(false, fH, false, bHN, (short)0, aHN, false, false);
      }

      // GRU gate epilogue straight out of the accumulators.
      // Acc layout: lane N = lane&15, VGPR v -> M = v + 8*(lane>>4).
      const int   half = lane >> 4;
      const int   j    = n0 + (lane & 15);
      const float bIRs = b_ih[j],       bHRs = b_hh[j];
      const float bIZs = b_ih[j + 256], bHZs = b_hh[j + 256];
      const float bINs = b_ih[j + 512], bHNs = b_hh[j + 512];
      #pragma unroll
      for (int v = 0; v < 8; ++v) {
        const int   m    = m0 + v + 8 * half;
        const float hold = hf[cur][m * HID + j];
        const float g    = G[(size_t)(row0 + m) * SEQ + s];
        const float r    = sigmoid_fast(aIR[v] + bIRs + aHR[v] + bHRs);
        const float z    = sigmoid_fast(aIZ[v] + bIZs + aHZ[v] + bHZs);
        const float n    = tanhf(aIN[v] + bINs + r * (aHN[v] + bHNs));
        const float hnew = (1.0f - z) * n + z * hold;
        const float hout = g * hnew + (1.0f - g) * hold;
        hf[nxt][m * HID + j] = hout;
        hb[nxt][m * HID + j] = (bf16_t)hout;
      }
    }
    __syncthreads();
    cur ^= 1;
  }

  // Emit final h (B, 1, HID) in fp32.
  for (int i = 0; i < M_TILE; ++i) {
    out[(size_t)(row0 + i) * HID + tid] = hf[cur][i * HID + tid];
  }
}

// ---------------------------------------------------------------------------
// Harness entry point.
// ---------------------------------------------------------------------------
extern "C" void kernel_launch(void* const* d_in, const int* in_sizes, int n_in,
                              void* d_out, int out_size, void* d_ws, size_t ws_size,
                              hipStream_t stream) {
  const float* C   = (const float*)d_in[0];
  const float* G   = (const float*)d_in[1];
  const float* Wih = (const float*)d_in[2];
  const float* Whh = (const float*)d_in[3];
  const float* bih = (const float*)d_in[4];
  const float* bhh = (const float*)d_in[5];
  float* out = (float*)d_out;

  bf16_t* WihB = (bf16_t*)d_ws;
  bf16_t* WhhB = WihB + (size_t)G3 * HID;

  const int nW = G3 * HID;
  convert_weights_kernel<<<(nW + 255) / 256, 256, 0, stream>>>(Wih, Whh, WihB, WhhB);
  gru_scan_kernel<<<B_TOTAL / M_TILE, THREADS, 0, stream>>>(C, G, WihB, WhhB, bih, bhh, out);
}
